// MoELayer_28132035789377
// MI455X (gfx1250) — matrix operand order repaired and checked by reading, not compiled, not run on hardware
//
#include <hip/hip_runtime.h>
#include <math.h>

// ---------------------------------------------------------------------------
// MoE conv layer for MI455X (gfx1250, wave32, WMMA).
// out = x + sum_e gate_w[b,e] * k[b,c] * ReLU(conv3x3_e(x) + bias_e[c])
// Only the top-2 experts have nonzero gate weight -> compute only those.
// Conv done as implicit GEMM with v_wmma_f32_16x16x32_bf16 (bf16 in, f32 acc).
// ---------------------------------------------------------------------------

typedef __attribute__((ext_vector_type(16))) __bf16 v16bf;
typedef __attribute__((ext_vector_type(8)))  float  v8f;

#define B_DIM 16
#define C_DIM 64
#define H_DIM 128
#define W_DIM 128
#define NEXP  4

// LDS geometry (elements are bf16)
#define CI_STR 72                                  // padded K stride (mult of 8 for b128 align, odd dword phase for banks)
#define A_COLS 130                                 // 128 pixels + 2 halo columns
#define LDS_A_ELEMS (3 * A_COLS * CI_STR)          // 28080
#define LDS_B_ELEMS (9 * C_DIM * CI_STR)           // 41472
#define LDS_BYTES   ((LDS_A_ELEMS + LDS_B_ELEMS) * 2)   // 139104 B (<160KB -> 2 blocks/WGP)

union Frag {
    uint4 u[2];
    v16bf v;
};

// ---------------------------------------------------------------------------
// Kernel 1: gate. One block per batch image. pooled mean -> linear -> softmax
// -> top-2 dense mask. Writes 4 floats per image into ws.
// ---------------------------------------------------------------------------
__global__ void gate_kernel(const float* __restrict__ x,
                            const float* __restrict__ gw,
                            const float* __restrict__ gb,
                            float* __restrict__ gate_out) {
    __shared__ float red[256];
    __shared__ float pooled[C_DIM];
    __shared__ float lg[NEXP];
    const int b = blockIdx.x, tid = threadIdx.x;

    for (int c = 0; c < C_DIM; ++c) {
        const float* plane = x + ((size_t)b * C_DIM + c) * (H_DIM * W_DIM);
        float s = 0.f;
        for (int j = tid; j < H_DIM * W_DIM; j += 256) s += plane[j];
        red[tid] = s;
        __syncthreads();
        for (int off = 128; off > 0; off >>= 1) {
            if (tid < off) red[tid] += red[tid + off];
            __syncthreads();
        }
        if (tid == 0) pooled[c] = red[0] * (1.f / (H_DIM * W_DIM));
        __syncthreads();
    }
    if (tid < NEXP) {
        float l = gb[tid];
        for (int c = 0; c < C_DIM; ++c) l += pooled[c] * gw[tid * C_DIM + c];
        lg[tid] = l;
    }
    __syncthreads();
    if (tid == 0) {
        float m = lg[0];
        for (int e = 1; e < NEXP; ++e) m = fmaxf(m, lg[e]);
        float w[NEXP], s = 0.f;
        for (int e = 0; e < NEXP; ++e) { w[e] = expf(lg[e] - m); s += w[e]; }
        for (int e = 0; e < NEXP; ++e) w[e] /= s;
        int i1 = 0;
        for (int e = 1; e < NEXP; ++e) if (w[e] > w[i1]) i1 = e;     // stable argmax
        int i2 = (i1 == 0) ? 1 : 0;
        for (int e = 0; e < NEXP; ++e) if (e != i1 && w[e] > w[i2]) i2 = e;
        for (int e = 0; e < NEXP; ++e)
            gate_out[b * NEXP + e] = (e == i1 || e == i2) ? w[e] : 0.f;
    }
}

// ---------------------------------------------------------------------------
// Kernel 2: repack conv weights f32 [E][O][I][3][3] -> bf16 [E][tap][O][I]
// (K = I contiguous so WMMA B fragments are contiguous 16B runs).
// ---------------------------------------------------------------------------
__global__ void wcvt_kernel(const float* __restrict__ cw,
                            __bf16* __restrict__ wbf) {
    const int i   = blockIdx.x * 256 + threadIdx.x;     // out index, coalesced write
    const int ci  = i & 63;
    const int n   = (i >> 6) & 63;
    const int tap = (i >> 12) % 9;
    const int e   = (i >> 12) / 9;
    wbf[i] = (__bf16)cw[(((e * C_DIM + n) * C_DIM + ci) * 9) + tap];
}

// ---------------------------------------------------------------------------
// Kernel 3: implicit-GEMM conv + gated epilogue. One block per (b, h):
// M = 128 pixels of row h, N = 64 out channels, K = 64 ci x 9 taps.
// 8 waves, each wave owns 16 pixels x 64 channels (4 WMMA tiles).
// ---------------------------------------------------------------------------
__global__ __launch_bounds__(256)
void moe_conv_kernel(const float* __restrict__ x,
                     const float* __restrict__ kvec,
                     const float* __restrict__ conv_b,
                     const float* __restrict__ gate,
                     const __bf16* __restrict__ wbf,
                     float* __restrict__ out) {
    extern __shared__ char smem[];
    __bf16* lds_a = (__bf16*)smem;                         // [3][130][CI_STR]
    __bf16* lds_b = (__bf16*)(smem + LDS_A_ELEMS * 2);     // [9][64][CI_STR]

    const int tid = threadIdx.x;
    const int b = blockIdx.x / H_DIM;
    const int h = blockIdx.x % H_DIM;

    // ---- Stage A: rows h-1..h+1, all 64 ch, halo cols zero-padded, f32->bf16
    for (int i = tid; i < 3 * C_DIM * A_COLS; i += 256) {
        const int col = i % A_COLS;            // consecutive i -> consecutive w (coalesced)
        const int rc  = i / A_COLS;
        const int ci  = rc % C_DIM;
        const int r   = rc / C_DIM;
        const int hh  = h + r - 1;
        const int ww  = col - 1;
        float v = 0.f;
        if (hh >= 0 && hh < H_DIM && ww >= 0 && ww < W_DIM)
            v = x[(((size_t)b * C_DIM + ci) * H_DIM + hh) * W_DIM + ww];
        lds_a[(r * A_COLS + col) * CI_STR + ci] = (__bf16)v;
    }

    const int lane = tid & 31;
    const int half = lane >> 4;       // lane 16..31 -> K offsets 8..15, 24..31 of each 32-K chunk (A and B)
    const int nl   = lane & 15;       // A: M row within tile; B/D: column N
    const int wave = tid >> 5;
    const int wb   = wave * 16;       // this wave's pixel base

    float kk[4];
    #pragma unroll
    for (int t = 0; t < 4; ++t) kk[t] = kvec[b * C_DIM + t * 16 + nl];

    const v8f vzero = {0.f, 0.f, 0.f, 0.f, 0.f, 0.f, 0.f, 0.f};
    v8f oacc[4];
    #pragma unroll
    for (int t = 0; t < 4; ++t) oacc[t] = vzero;

    for (int e = 0; e < NEXP; ++e) {
        // force wave-uniform gate value -> scalar branch, EXEC stays all-ones for WMMA
        const float ge = __int_as_float(
            __builtin_amdgcn_readfirstlane(__float_as_int(gate[b * NEXP + e])));
        if (ge == 0.f) continue;                 // block-uniform skip of unselected experts

        __syncthreads();                         // prior expert done reading lds_b
        // ---- Stage B: expert weights bf16 [tap][n][ci] -> LDS, b128 chunks
        for (int i = tid; i < 9 * C_DIM * 8; i += 256) {
            const int tn = i >> 3;               // tap*64 + n
            const int cc = (i & 7) << 3;         // ci chunk base
            const uint4 vv = *(const uint4*)(wbf + ((size_t)e * (9 * C_DIM) + tn) * C_DIM + cc);
            *(uint4*)(lds_b + tn * CI_STR + cc) = vv;
        }
        __syncthreads();

        v8f acc[4];
        #pragma unroll
        for (int t = 0; t < 4; ++t) acc[t] = vzero;

        #pragma unroll
        for (int tap = 0; tap < 9; ++tap) {
            const int kh = tap / 3, kw = tap % 3;
            // A row for this lane's M: pixel (wb+nl), shifted by kw (halo offset folded in)
            const int arow = (kh * A_COLS + (wb + nl + kw)) * CI_STR;
            #pragma unroll
            for (int kc = 0; kc < C_DIM; kc += 32) {
                // A 16x32 bf16 per-lane layout: two 8-elt runs at K = kc+8*half, kc+16+8*half
                Frag af;
                af.u[0] = *(const uint4*)(lds_a + arow + kc + 8 * half);
                af.u[1] = *(const uint4*)(lds_a + arow + kc + 16 + 8 * half);
                // Batch-load ALL four B fragments first so up to 10 ds_loads are in
                // flight before the first WMMA -> descending s_wait_dscnt instead of
                // a hard wait-0 stall per WMMA.
                Frag bf[4];
                #pragma unroll
                for (int t = 0; t < 4; ++t) {
                    // B 32x16 bf16 per-lane layout: two 8-elt runs at K = kc+8*half, kc+16+8*half, col N=nl
                    const int boff = (tap * C_DIM + t * 16 + nl) * CI_STR + kc + 8 * half;
                    bf[t].u[0] = *(const uint4*)(lds_b + boff);
                    bf[t].u[1] = *(const uint4*)(lds_b + boff + 16);
                }
                #pragma unroll
                for (int t = 0; t < 4; ++t) {
                    acc[t] = __builtin_amdgcn_wmma_f32_16x16x32_bf16(
                        false, af.v, false, bf[t].v, (short)0, acc[t], false, false);
                }
            }
        }

        // ---- epilogue: bias -> ReLU -> * (gate * k), accumulate across experts
        #pragma unroll
        for (int t = 0; t < 4; ++t) {
            const int c = t * 16 + nl;
            const float bias = conv_b[e * C_DIM + c];
            const float sc = ge * kk[t];
            #pragma unroll
            for (int v = 0; v < 8; ++v) {
                float f = acc[t][v] + bias;
                f = f > 0.f ? f : 0.f;
                oacc[t][v] += f * sc;
            }
        }
    }

    // ---- residual add + store (D layout: VGPR v -> pixel wb + 8*half + v, lane -> channel)
    #pragma unroll
    for (int t = 0; t < 4; ++t) {
        const int c = t * 16 + nl;
        const size_t base = (((size_t)b * C_DIM + c) * H_DIM + h) * W_DIM + wb + 8 * half;
        const float4 x0 = *(const float4*)(x + base);
        const float4 x1 = *(const float4*)(x + base + 4);
        float4 o0 = make_float4(x0.x + oacc[t][0], x0.y + oacc[t][1],
                                x0.z + oacc[t][2], x0.w + oacc[t][3]);
        float4 o1 = make_float4(x1.x + oacc[t][4], x1.y + oacc[t][5],
                                x1.z + oacc[t][6], x1.w + oacc[t][7]);
        *(float4*)(out + base) = o0;
        *(float4*)(out + base + 4) = o1;
    }
}

// ---------------------------------------------------------------------------
// Host launcher. ws layout: [0,256) gate weights (64 f32), [256, 256+294912)
// bf16 repacked conv weights. Total ws use ~295 KB.
// ---------------------------------------------------------------------------
extern "C" void kernel_launch(void* const* d_in, const int* in_sizes, int n_in,
                              void* d_out, int out_size, void* d_ws, size_t ws_size,
                              hipStream_t stream) {
    const float* x  = (const float*)d_in[0];   // [16,64,128,128]
    const float* kv = (const float*)d_in[1];   // [16,64,1,1]
    const float* gw = (const float*)d_in[2];   // [4,64]
    const float* gb = (const float*)d_in[3];   // [4]
    const float* cw = (const float*)d_in[4];   // [4,64,64,3,3]
    const float* cb = (const float*)d_in[5];   // [4,64]
    float* out = (float*)d_out;

    float*  gate_ws = (float*)d_ws;
    __bf16* wbf     = (__bf16*)((char*)d_ws + 256);

    gate_kernel<<<B_DIM, 256, 0, stream>>>(x, gw, gb, gate_ws);

    const int wtotal = NEXP * C_DIM * C_DIM * 9;          // 147456
    wcvt_kernel<<<wtotal / 256, 256, 0, stream>>>(cw, wbf);

    moe_conv_kernel<<<B_DIM * H_DIM, 256, LDS_BYTES, stream>>>(
        x, kv, cb, gate_ws, wbf, out);
}
